// GATPolicy_12343736009369
// MI455X (gfx1250) — matrix operand order repaired and checked by reading, not compiled, not run on hardware
//
#include <hip/hip_runtime.h>
#include <hip/hip_fp16.h>

typedef __attribute__((ext_vector_type(16))) _Float16 v16h;
typedef __attribute__((ext_vector_type(8)))  _Float16 v8h;
typedef __attribute__((ext_vector_type(8)))  float    v8f;

#define DIM 256
#define NT  16   // 256 / 16 column tiles
#define KT  8    // 256 / 32 k-steps
#define BM  2    // m-tiles per wave
#define BN  4    // n-tiles per wave

// ---- order-preserving float<->uint encoding for atomicMax on floats ----
__device__ __forceinline__ unsigned fenc(float f) {
    unsigned u = __float_as_uint(f);
    return (u & 0x80000000u) ? ~u : (u | 0x80000000u);
}
__device__ __forceinline__ float fdec(unsigned u) {
    u = (u & 0x80000000u) ? (u & 0x7FFFFFFFu) : ~u;
    return __uint_as_float(u);
}

__global__ void k_zero_f32(float* p, int n) {
    int i = blockIdx.x * 256 + threadIdx.x;
    if (i < n) p[i] = 0.f;
}

__global__ void k_init_max(unsigned* p, int n) {
    int i = blockIdx.x * 256 + threadIdx.x;
    if (i < n) p[i] = 0x007FFFFFu;   // fenc(-inf)
}

__global__ void k_f32_to_f16(const float* __restrict__ x, _Float16* __restrict__ y, int n) {
    int i = blockIdx.x * 256 + threadIdx.x;
    if (i < n) y[i] = (_Float16)x[i];
}

// Pre-swizzle W (row-major KxN f32) into the per-lane B layout of
// v_wmma_f32_16x16x32_f16: element j of lane's v16h = W[k0 + (lane<16?0:8) + (j&7) + (j>=8?16:0)][tn*16 + lane&15]
__global__ void k_pack_w(const float* __restrict__ W, _Float16* __restrict__ Wp) {
    int idx = blockIdx.x * 256 + threadIdx.x;     // 65536 total
    if (idx >= DIM * DIM) return;
    int j    = idx & 15;
    int lane = (idx >> 4) & 31;
    int kt   = (idx >> 9) & 7;
    int tn   = idx >> 12;
    int k = kt * 32 + ((lane < 16) ? 0 : 8) + (j & 7) + ((j >> 3) << 4);
    int n = tn * 16 + (lane & 15);
    Wp[idx] = (_Float16)W[k * DIM + n];
}

// H[m][n] = sum_k A[m][k] * W[k][n]
// Register-blocked 2x4: one wave computes a 32x64 output region (8 WMMA tiles),
// reusing 2 A-fragments x 4 B-fragments per k-step (8 WMMAs per 8 loads).
// Base pointers hoisted; all k/j offsets are compile-time immediates.
__global__ void k_gemm_wmma(const _Float16* __restrict__ A, const _Float16* __restrict__ Wp,
                            float* __restrict__ Hout, int N, int mtiles) {
    int lane = threadIdx.x & 31;
    int wid  = threadIdx.x >> 5;
    int mg   = (mtiles + BM - 1) / BM;
    int job  = blockIdx.x * 8 + wid;
    if (job >= mg * (NT / BN)) return;
    int tmb = (job / (NT / BN)) * BM;
    int tnb = (job % (NT / BN)) * BN;

    int ksel = (lane >> 4) * 8;   // A: lanes 0-15 hold K{0..7,16..23}, lanes 16-31 K{8..15,24..31}
    int lrow = lane & 15;

    const _Float16* ap[BM];
#pragma unroll
    for (int i = 0; i < BM; ++i) {
        int row = (tmb + i) * 16 + lrow;
        if (row >= N) row = N - 1;          // tail group: clamp (stores are guarded)
        ap[i] = A + (size_t)row * DIM + ksel;
    }
    const _Float16* bp = Wp + (size_t)(tnb * KT * 32 + lane) * 16;

    v8f acc[BM][BN];
    v8f zero = {};
#pragma unroll
    for (int i = 0; i < BM; ++i)
#pragma unroll
        for (int j = 0; j < BN; ++j) acc[i][j] = zero;

#pragma unroll
    for (int kt = 0; kt < KT; ++kt) {
        v16h a[BM], b[BN];
#pragma unroll
        for (int i = 0; i < BM; ++i) {
            v8h alo = *(const v8h*)(ap[i] + kt * 32);        // imm offset
            v8h ahi = *(const v8h*)(ap[i] + kt * 32 + 16);   // imm offset
            a[i] = __builtin_shufflevector(alo, ahi, 0, 1, 2, 3, 4, 5, 6, 7,
                                                     8, 9, 10, 11, 12, 13, 14, 15);
        }
#pragma unroll
        for (int j = 0; j < BN; ++j)
            b[j] = *(const v16h*)(bp + kt * (32 * 16) + j * (KT * 32 * 16));  // imm offsets
#pragma unroll
        for (int i = 0; i < BM; ++i)
#pragma unroll
            for (int j = 0; j < BN; ++j)
                acc[i][j] = __builtin_amdgcn_wmma_f32_16x16x32_f16(
                    false, a[i], false, b[j], (short)0, acc[i][j], false, false);
    }

    // C/D layout: VGPR r -> row (lane<16 ? r : 8+r), col = lane & 15
    int rsel = (lane < 16) ? 0 : 8;
#pragma unroll
    for (int i = 0; i < BM; ++i) {
        int rbase = (tmb + i) * 16 + rsel;
#pragma unroll
        for (int j = 0; j < BN; ++j) {
            int col = (tnb + j) * 16 + lrow;
#pragma unroll
            for (int r = 0; r < 8; ++r) {
                int row = rbase + r;
                if (row < N) Hout[(size_t)row * DIM + col] = acc[i][j][r];
            }
        }
    }
}

// al_s[n] = H[n]·a_src, al_d[n] = H[n]·a_dst ; one wave32 per node
__global__ void k_logits(const float* __restrict__ H, const float* __restrict__ as_,
                         const float* __restrict__ ad_, float* __restrict__ alS,
                         float* __restrict__ alD, int n) {
    int lane = threadIdx.x & 31;
    int node = (blockIdx.x * blockDim.x + threadIdx.x) >> 5;
    if (node >= n) return;
    const float* row = H + (size_t)node * DIM;
    float s = 0.f, d = 0.f;
#pragma unroll
    for (int i = 0; i < 8; ++i) {
        int c = lane * 8 + i;
        float v = row[c];
        s += v * as_[c];
        d += v * ad_[c];
    }
#pragma unroll
    for (int off = 16; off >= 1; off >>= 1) {
        s += __shfl_xor(s, off, 32);
        d += __shfl_xor(d, off, 32);
    }
    if (lane == 0) { alS[node] = s; alD[node] = d; }
}

// e = leaky_relu(al_s[src]+al_d[dst]); segment max over dst via encoded atomicMax
__global__ void k_edge_max(const int* __restrict__ eidx, int E, int ET,
                           const float* __restrict__ alS, const float* __restrict__ alD,
                           float* __restrict__ ebuf, unsigned* __restrict__ menc) {
    int j = blockIdx.x * 256 + threadIdx.x;
    if (j >= ET) return;
    int s, d;
    if (j < E) { s = eidx[j]; d = eidx[E + j]; } else { s = d = j - E; }
    float v = alS[s] + alD[d];
    v = (v > 0.f) ? v : 0.2f * v;
    ebuf[j] = v;
    atomicMax(&menc[d], fenc(v));
}

// ex = exp(e - m[dst]); segment sum over dst
__global__ void k_edge_exp(const int* __restrict__ eidx, int E, int ET,
                           float* __restrict__ ebuf, const unsigned* __restrict__ menc,
                           float* __restrict__ den) {
    int j = blockIdx.x * 256 + threadIdx.x;
    if (j >= ET) return;
    int d = (j < E) ? eidx[E + j] : (j - E);
    float x = expf(ebuf[j] - fdec(menc[d]));
    ebuf[j] = x;
    atomicAdd(&den[d], x);
}

// acc[dst] += h[src] * alpha ; one wave32 per edge, 8 floats/lane (L2-resident atomics)
__global__ void k_scatter(const int* __restrict__ eidx, int E, int ET,
                          const float* __restrict__ ebuf, const float* __restrict__ den,
                          const float* __restrict__ H, float* __restrict__ acc) {
    int lane = threadIdx.x & 31;
    int j = (blockIdx.x * blockDim.x + threadIdx.x) >> 5;
    if (j >= ET) return;
    int s, d;
    if (j < E) { s = eidx[j]; d = eidx[E + j]; } else { s = d = j - E; }
    float alpha = ebuf[j] / den[d];
    const float4* hrow = (const float4*)(H + (size_t)s * DIM) + lane * 2;
    float* arow = acc + (size_t)d * DIM + lane * 8;
    float4 h0 = hrow[0], h1 = hrow[1];
    atomicAdd(arow + 0, h0.x * alpha);
    atomicAdd(arow + 1, h0.y * alpha);
    atomicAdd(arow + 2, h0.z * alpha);
    atomicAdd(arow + 3, h0.w * alpha);
    atomicAdd(arow + 4, h1.x * alpha);
    atomicAdd(arow + 5, h1.y * alpha);
    atomicAdd(arow + 6, h1.z * alpha);
    atomicAdd(arow + 7, h1.w * alpha);
}

// layer-1 epilogue: relu(acc + b) -> f16 (input of layer-2 GEMM)
__global__ void k_bias_relu_f16(const float* __restrict__ acc, const float* __restrict__ b,
                                _Float16* __restrict__ out, int n) {
    int i = blockIdx.x * 256 + threadIdx.x;
    if (i < n) out[i] = (_Float16)fmaxf(acc[i] + b[i & 255], 0.f);
}

// layer-2 epilogue: in-place relu(d_out + b)
__global__ void k_bias_relu_f32(float* __restrict__ io, const float* __restrict__ b, int n) {
    int i = blockIdx.x * 256 + threadIdx.x;
    if (i < n) io[i] = fmaxf(io[i] + b[i & 255], 0.f);
}

extern "C" void kernel_launch(void* const* d_in, const int* in_sizes, int n_in,
                              void* d_out, int out_size, void* d_ws, size_t ws_size,
                              hipStream_t stream) {
    const float* x    = (const float*)d_in[0];
    const int*   eidx = (const int*)d_in[1];   // harness convention: integer -> const int*
    const float* W1   = (const float*)d_in[2];
    const float* a1s  = (const float*)d_in[3];
    const float* a1d  = (const float*)d_in[4];
    const float* b1   = (const float*)d_in[5];
    const float* W2   = (const float*)d_in[6];
    const float* a2s  = (const float*)d_in[7];
    const float* a2d  = (const float*)d_in[8];
    const float* b2   = (const float*)d_in[9];
    (void)n_in; (void)out_size; (void)ws_size;

    const int N  = in_sizes[0] / DIM;   // 50000
    const int E  = in_sizes[1] / 2;     // 800000
    const int ET = E + N;               // + self loops
    const int NE = N * DIM;
    const int mtiles = N / 16;          // 3125 exactly

    char* p = (char*)d_ws;
    auto alloc = [&](size_t bytes) -> char* {
        char* r = p;
        p += (bytes + 255) & ~(size_t)255;
        return r;
    };
    _Float16* Xh  = (_Float16*)alloc((size_t)NE * 2);   // layer input, f16
    float*    H   = (float*)   alloc((size_t)NE * 4);   // h = x@W (f32)
    float*    ACC = (float*)   alloc((size_t)NE * 4);   // layer-1 aggregation
    _Float16* Wp1 = (_Float16*)alloc((size_t)DIM * DIM * 2);
    _Float16* Wp2 = (_Float16*)alloc((size_t)DIM * DIM * 2);
    float*    alS = (float*)   alloc((size_t)N * 4);
    float*    alD = (float*)   alloc((size_t)N * 4);
    unsigned* men = (unsigned*)alloc((size_t)N * 4);
    float*    den = (float*)   alloc((size_t)N * 4);
    float*    ebuf = (float*)  alloc((size_t)ET * 4);

    dim3 blk(256);
    int gNE = (NE + 255) / 256;
    int gN  = (N + 255) / 256;
    int gET = (ET + 255) / 256;
    int gWaveN  = (N * 32 + 255) / 256;
    int gWaveET = (int)(((size_t)ET * 32 + 255) / 256);
    int gPack   = (DIM * DIM + 255) / 256;
    int mg      = (mtiles + BM - 1) / BM;
    int jobs    = mg * (NT / BN);
    int gGemm   = (jobs + 7) / 8;

    k_f32_to_f16<<<gNE, blk, 0, stream>>>(x, Xh, NE);
    k_pack_w<<<gPack, blk, 0, stream>>>(W1, Wp1);
    k_pack_w<<<gPack, blk, 0, stream>>>(W2, Wp2);

    for (int layer = 0; layer < 2; ++layer) {
        const _Float16* Wp = layer ? Wp2 : Wp1;
        const float* as_ = layer ? a2s : a1s;
        const float* ad_ = layer ? a2d : a1d;
        const float* b_  = layer ? b2  : b1;
        float* acc = layer ? (float*)d_out : ACC;

        k_zero_f32<<<gNE, blk, 0, stream>>>(acc, NE);
        k_zero_f32<<<gN, blk, 0, stream>>>(den, N);
        k_init_max<<<gN, blk, 0, stream>>>(men, N);

        k_gemm_wmma<<<gGemm, blk, 0, stream>>>(Xh, Wp, H, N, mtiles);
        k_logits<<<gWaveN, blk, 0, stream>>>(H, as_, ad_, alS, alD, N);
        k_edge_max<<<gET, blk, 0, stream>>>(eidx, E, ET, alS, alD, ebuf, men);
        k_edge_exp<<<gET, blk, 0, stream>>>(eidx, E, ET, ebuf, men, den);
        k_scatter<<<gWaveET, blk, 0, stream>>>(eidx, E, ET, ebuf, den, H, acc);

        if (layer == 0)
            k_bias_relu_f16<<<gNE, blk, 0, stream>>>(acc, b_, Xh, NE);  // Xh := layer-2 input
        else
            k_bias_relu_f32<<<gNE, blk, 0, stream>>>((float*)d_out, b_, NE);
    }
}
